// GroupQueryAttention_14508399526711
// MI455X (gfx1250) — compile-verified, matrix operand order checked by reading, not run on hardware
//
#include <hip/hip_runtime.h>
#include <hip/hip_bf16.h>
#include <stdint.h>

// ---------------- problem constants (from reference) ----------------
static constexpr int kB   = 2;
static constexpr int kS   = 2048;
static constexpr int kD   = 2048;
static constexpr int kH   = 32;
static constexpr int kG   = 8;
static constexpr int kHD  = 64;              // head dim
static constexpr int kGS  = kH / kG;         // 4 query heads per kv group
static constexpr int kM   = kB * kS;         // 4096 rows
static constexpr int kNKV = kG * kHD;        // 512

typedef unsigned short u16;
typedef unsigned int   u32;
typedef unsigned long long u64;
typedef __attribute__((ext_vector_type(16))) __bf16 v16bf;
typedef __attribute__((ext_vector_type(8)))  float  v8f;

__device__ __forceinline__ u16 f2bf(float f) {
  u32 u = __float_as_uint(f);
  u32 r = u + 0x7FFFu + ((u >> 16) & 1u);   // round-to-nearest-even
  return (u16)(r >> 16);
}

union Frag {
  uint4 q[2];
  v16bf v;
  u16   s[16];
};

// 32-byte fragment load as two 16B chunks (lowers to ds_load_b128)
__device__ __forceinline__ v16bf load_frag(const u16* p) {
  Frag f;
  f.q[0] = *(const uint4*)(p);
  f.q[1] = *(const uint4*)(p + 8);
  return f.v;
}

__device__ __forceinline__ v8f wmma_bf16(v16bf a, v16bf b, v8f c) {
  return __builtin_amdgcn_wmma_f32_16x16x32_bf16(
      /*neg_a=*/false, a, /*neg_b=*/false, b,
      /*c_mod=*/(short)0, c, /*reuse_a=*/false, /*reuse_b=*/false);
}

// ---- CDNA5 async memory->LDS copy (ASYNCcnt path, no VGPR round-trip) ----
__device__ __forceinline__ void async_load_b128(const u16* gptr, u16* lptr) {
  u32 lds = (u32)(uintptr_t)lptr;     // flat aperture: LDS address is the low 32 bits of the pointer
  u64 ga  = (u64)(uintptr_t)gptr;
  asm volatile("global_load_async_to_lds_b128 %0, %1, off"
               :: "v"(lds), "v"(ga) : "memory");
}
__device__ __forceinline__ void wait_async0() {
  asm volatile("s_wait_asynccnt 0x0" ::: "memory");
}

// ---------------- fp32 -> bf16 conversion (straight) ----------------
__global__ void cvt_f32_bf16(const float* __restrict__ in, u16* __restrict__ out, int n4) {
  int i = blockIdx.x * blockDim.x + threadIdx.x;
  int stride = gridDim.x * blockDim.x;
  const float4* in4 = (const float4*)in;
  u32* o = (u32*)out;
  for (int j = i; j < n4; j += stride) {
    float4 f = in4[j];
    o[2 * j]     = (u32)f2bf(f.x) | ((u32)f2bf(f.y) << 16);
    o[2 * j + 1] = (u32)f2bf(f.z) | ((u32)f2bf(f.w) << 16);
  }
}

// ---------------- fp32 -> bf16 conversion (transposing) ----------------
// in: [R][C] f32 row-major -> out: [C][R] bf16 row-major. 32x32 LDS tiles.
__global__ __launch_bounds__(256) void cvt_f32_bf16_t(
    const float* __restrict__ in, u16* __restrict__ out, int R, int C) {
  __shared__ float t[32][33];
  const int bc = blockIdx.x * 32, br = blockIdx.y * 32;
  const int tx = threadIdx.x, ty = threadIdx.y;   // (32, 8)
#pragma unroll
  for (int i = 0; i < 32; i += 8)
    t[ty + i][tx] = in[(size_t)(br + ty + i) * C + bc + tx];
  __syncthreads();
#pragma unroll
  for (int i = 0; i < 32; i += 8)
    out[(size_t)(bc + ty + i) * R + br + tx] = f2bf(t[tx][ty + i]);
}

// ---------------- bf16 GEMM: C = A[M,K] @ B[K,N] + bias ----------------
// B is supplied PRE-TRANSPOSED: Btg[N][K] (WMMA-B layout), so both tiles
// stage via async memory->LDS. 128x128 block tile, BK=32, 8 waves in
// 4(M) x 2(N), each wave 32x64 via 2x4 WMMA 16x16x32 accumulators.
// LDS rows padded to 40 elems (80B): the 16 fragment rows hit distinct banks.
static constexpr int AS = 40;  // LDS row stride in elements (32 + 8 pad)

__global__ __launch_bounds__(256) void gemm_bf16(
    const u16* __restrict__ A, const u16* __restrict__ Btg,
    const float* __restrict__ bias, u16* __restrict__ outb,
    float* __restrict__ outf, u16* __restrict__ outbt,
    int Mdim, int Ndim, int Kdim) {
  __shared__ u16 As[128 * AS];   // A tile, [m][k]
  __shared__ u16 Bs[128 * AS];   // B tile, [n][k]

  const int tid  = threadIdx.x;
  const int lane = tid & 31;
  const int ln   = lane & 15;
  const int hl   = lane >> 4;
  const int w    = tid >> 5;
  const int wm   = w >> 1;       // 0..3
  const int wn   = w & 1;        // 0..1
  const int bm   = blockIdx.y * 128;
  const int bn   = blockIdx.x * 128;

  v8f zero = {};
  v8f acc[2][4];
#pragma unroll
  for (int i = 0; i < 2; ++i)
#pragma unroll
    for (int n = 0; n < 4; ++n) acc[i][n] = zero;

  // per-thread tile-load coordinates (both tiles are 128 rows x 32 cols)
  const int tr = tid >> 1;                 // 0..127 tile row
  const int tc = (tid & 1) * 16;           // 0 / 16 k chunk

  const u16* aSrc = A   + (size_t)(bm + tr) * Kdim + tc;
  const u16* bSrc = Btg + (size_t)(bn + tr) * Kdim + tc;
  u16* aDst = &As[tr * AS + tc];
  u16* bDst = &Bs[tr * AS + tc];

  for (int k0 = 0; k0 < Kdim; k0 += 32) {
    // ---- stage A and B tiles via async memory->LDS (ASYNCcnt) ----
    async_load_b128(aSrc + k0,     aDst);
    async_load_b128(aSrc + k0 + 8, aDst + 8);
    async_load_b128(bSrc + k0,     bDst);
    async_load_b128(bSrc + k0 + 8, bDst + 8);
    if (k0 + 32 < Kdim) {
      __builtin_prefetch(aSrc + k0 + 32, 0, 1);
      __builtin_prefetch(bSrc + k0 + 32, 0, 1);
    }
    wait_async0();
    __syncthreads();

    // ---- WMMA on the tile ----
    const u16* aP = &As[(wm * 32 + ln) * AS + hl * 16];
    v16bf af0 = load_frag(aP);
    v16bf af1 = load_frag(aP + 16 * AS);
    const u16* bP = &Bs[(wn * 64 + ln) * AS + hl * 16];
#pragma unroll
    for (int n = 0; n < 4; ++n) {
      v16bf bf = load_frag(bP + n * 16 * AS);
      acc[0][n] = wmma_bf16(af0, bf, acc[0][n]);
      acc[1][n] = wmma_bf16(af1, bf, acc[1][n]);
    }
    __syncthreads();
  }

  // ---- epilogue: bias add; store bf16, f32, and/or transposed bf16 ----
#pragma unroll
  for (int i = 0; i < 2; ++i)
#pragma unroll
    for (int n = 0; n < 4; ++n)
#pragma unroll
      for (int r = 0; r < 8; ++r) {
        int gm = bm + wm * 32 + i * 16 + r + 8 * hl;
        int gn = bn + wn * 64 + n * 16 + ln;
        float v = acc[i][n][r] + bias[gn];
        if (outb)  outb[(size_t)gm * Ndim + gn] = f2bf(v);
        if (outf)  outf[(size_t)gm * Ndim + gn] = v;
        if (outbt) outbt[(size_t)gn * Mdim + gm] = f2bf(v);  // C^T (for V)
      }
}

// ---------------- fused causal flash attention ----------------
// Grid: (S/64, H, B). Block: 128 threads = 4 waves; each wave owns 16 query
// rows, iterates 64-wide causal key tiles. K stages as [key][dim] (natural
// WMMA-B layout for Q@K^T); V^T is precomputed globally, so the [dim][key]
// tile for P@V also stages with straight async copies.
static constexpr int KS = 72;  // LDS row stride (64 + 8 pad)

__global__ __launch_bounds__(128) void flash_attn(
    const u16* __restrict__ Qb, const u16* __restrict__ Kb,
    const u16* __restrict__ VtG, u16* __restrict__ Ob) {
  __shared__ u16 Ks[64 * KS];
  __shared__ u16 Vt[64 * KS];
  __shared__ u16 Pl[4 * 16 * KS];   // per-wave P staging

  const int qt = blockIdx.x, h = blockIdx.y, b = blockIdx.z;
  const int g  = h / kGS;
  const int tid  = threadIdx.x;
  const int lane = tid & 31;
  const int ln   = lane & 15;
  const int hl   = lane >> 4;
  const int w    = tid >> 5;

  const int qrow0 = qt * 64 + w * 16;

  // Q fragments (16 rows x 64 dim = 2 A-fragments), loaded once
  v16bf qf[2];
  {
    const u16* qp = Qb + (size_t)(b * kS + qrow0 + ln) * kD + h * kHD + hl * 16;
    qf[0] = load_frag(qp);
    qf[1] = load_frag(qp + 32);
  }

  v8f zero = {};
  v8f of[4];
#pragma unroll
  for (int d = 0; d < 4; ++d) of[d] = zero;
  float m[8], l[8];
#pragma unroll
  for (int r = 0; r < 8; ++r) { m[r] = -1e30f; l[r] = 0.0f; }
  const float scale = 0.125f;  // 1/sqrt(64)

  // per-thread tile-staging coordinates
  const int tr = tid >> 1;            // 0..63 tile row (key for K, dim for V^T)
  const int tc = (tid & 1) * 32;      // 0 / 32 column chunk

  const u16* kSrc = Kb  + (size_t)(b * kS + tr) * kNKV + g * kHD + tc;   // + kt*64*kNKV
  const u16* vSrc = VtG + (size_t)(g * kHD + tr) * kM + b * kS + tc;     // + kt*64
  u16* kDst = &Ks[tr * KS + tc];
  u16* vDst = &Vt[tr * KS + tc];

  for (int kt = 0; kt <= qt; ++kt) {
    // ---- stage K [key][dim] and V^T [dim][key] tiles via async->LDS ----
    {
      const u16* s0 = kSrc + (size_t)kt * 64 * kNKV;
      async_load_b128(s0,      kDst);
      async_load_b128(s0 + 8,  kDst + 8);
      async_load_b128(s0 + 16, kDst + 16);
      async_load_b128(s0 + 24, kDst + 24);
      const u16* s1 = vSrc + (size_t)kt * 64;
      async_load_b128(s1,      vDst);
      async_load_b128(s1 + 8,  vDst + 8);
      async_load_b128(s1 + 16, vDst + 16);
      async_load_b128(s1 + 24, vDst + 24);
    }
    if (kt < qt) {
      __builtin_prefetch(kSrc + (size_t)(kt + 1) * 64 * kNKV, 0, 1);
      __builtin_prefetch(vSrc + (size_t)(kt + 1) * 64, 0, 1);
    }
    wait_async0();
    __syncthreads();

    // ---- S = Q @ K^T  (4 fragments over 64 keys) ----
    v8f sf[4];
    {
      const u16* kp = &Ks[ln * KS + hl * 16];
#pragma unroll
      for (int n = 0; n < 4; ++n) {
        sf[n] = zero;
        v16bf b0 = load_frag(kp + n * 16 * KS);
        v16bf b1 = load_frag(kp + n * 16 * KS + 32);
        sf[n] = wmma_bf16(qf[0], b0, sf[n]);
        sf[n] = wmma_bf16(qf[1], b1, sf[n]);
      }
    }

    // ---- online softmax update (rows M = r + 8*hl, cols N = n*16 + ln) ----
    float pv[4][8];
#pragma unroll
    for (int r = 0; r < 8; ++r) {
      const int qg = qrow0 + r + 8 * hl;
      float rm = -1e30f;
#pragma unroll
      for (int n = 0; n < 4; ++n) {
        float v = sf[n][r] * scale;
        int kg = kt * 64 + n * 16 + ln;
        v = (kg <= qg) ? v : -1e30f;    // causal mask
        pv[n][r] = v;
        rm = fmaxf(rm, v);
      }
      rm = fmaxf(rm, __shfl_xor(rm, 1, 32));
      rm = fmaxf(rm, __shfl_xor(rm, 2, 32));
      rm = fmaxf(rm, __shfl_xor(rm, 4, 32));
      rm = fmaxf(rm, __shfl_xor(rm, 8, 32));
      float mn = fmaxf(m[r], rm);
      float alpha = __expf(m[r] - mn);
      float rs = 0.0f;
#pragma unroll
      for (int n = 0; n < 4; ++n) {
        float p = __expf(pv[n][r] - mn);
        pv[n][r] = p;
        rs += p;
      }
      rs += __shfl_xor(rs, 1, 32);
      rs += __shfl_xor(rs, 2, 32);
      rs += __shfl_xor(rs, 4, 32);
      rs += __shfl_xor(rs, 8, 32);
      l[r] = l[r] * alpha + rs;
      m[r] = mn;
#pragma unroll
      for (int d = 0; d < 4; ++d) of[d][r] *= alpha;
    }

    // ---- stage P (bf16) into per-wave LDS, reload as A fragments ----
    u16* pw = &Pl[w * 16 * KS];
#pragma unroll
    for (int n = 0; n < 4; ++n)
#pragma unroll
      for (int r = 0; r < 8; ++r)
        pw[(r + 8 * hl) * KS + n * 16 + ln] = f2bf(pv[n][r]);
    // same-wave LDS store->load is in-order; compiler waits DScnt on the loads

    // ---- O += P @ V ----
    {
      const u16* pA = pw + ln * KS + hl * 16;
      v16bf pa0 = load_frag(pA);
      v16bf pa1 = load_frag(pA + 32);
      const u16* vp = &Vt[ln * KS + hl * 16];
#pragma unroll
      for (int d = 0; d < 4; ++d) {
        v16bf vb0 = load_frag(vp + d * 16 * KS);
        v16bf vb1 = load_frag(vp + d * 16 * KS + 32);
        of[d] = wmma_bf16(pa0, vb0, of[d]);
        of[d] = wmma_bf16(pa1, vb1, of[d]);
      }
    }
    __syncthreads();
  }

  // ---- epilogue: normalize, store bf16 into [B*S, D] at head offset ----
  u16* op = Ob + (size_t)(b * kS + qrow0) * kD + h * kHD;
#pragma unroll
  for (int d = 0; d < 4; ++d)
#pragma unroll
    for (int r = 0; r < 8; ++r) {
      float v = of[d][r] / l[r];
      op[(size_t)(r + 8 * hl) * kD + d * 16 + ln] = f2bf(v);
    }
}

// ---------------- host-side orchestration ----------------
extern "C" void kernel_launch(void* const* d_in, const int* in_sizes, int n_in,
                              void* d_out, int out_size, void* d_ws, size_t ws_size,
                              hipStream_t stream) {
  (void)in_sizes; (void)n_in; (void)out_size; (void)ws_size;
  const float* x  = (const float*)d_in[0];
  const float* kv = (const float*)d_in[1];
  const float* Wq = (const float*)d_in[2];
  const float* bq = (const float*)d_in[3];
  const float* Wk = (const float*)d_in[4];
  const float* bk = (const float*)d_in[5];
  const float* Wv = (const float*)d_in[6];
  const float* bv = (const float*)d_in[7];
  const float* Wo = (const float*)d_in[8];
  const float* bo = (const float*)d_in[9];
  float* out = (float*)d_out;

  u16* ws = (u16*)d_ws;
  size_t off = 0;
  auto alloc = [&](size_t n) { u16* p = ws + off; off += n; return p; };
  u16* xb   = alloc((size_t)kM * kD);
  u16* kvb  = alloc((size_t)kM * kD);
  u16* WqbT = alloc((size_t)kD * kD);     // [N][K]
  u16* WkbT = alloc((size_t)kNKV * kD);   // [N][K]
  u16* WvbT = alloc((size_t)kNKV * kD);   // [N][K]
  u16* WobT = alloc((size_t)kD * kD);     // [N][K]
  u16* Qb   = alloc((size_t)kM * kD);
  u16* Kb   = alloc((size_t)kM * kNKV);
  u16* VtG  = alloc((size_t)kNKV * kM);   // V^T: [dim][seq]
  u16* Ob   = alloc((size_t)kM * kD);

  auto cvt = [&](const float* in, u16* o, size_t n) {
    int n4 = (int)(n / 4);
    int blocks = (n4 + 255) / 256;
    if (blocks > 4096) blocks = 4096;
    cvt_f32_bf16<<<blocks, 256, 0, stream>>>(in, o, n4);
  };
  auto cvtT = [&](const float* in, u16* o, int R, int C) {  // [R][C] -> [C][R]
    dim3 gr(C / 32, R / 32), bl(32, 8);
    cvt_f32_bf16_t<<<gr, bl, 0, stream>>>(in, o, R, C);
  };
  cvt (x,  xb,  (size_t)kM * kD);
  cvt (kv, kvb, (size_t)kM * kD);
  cvtT(Wq, WqbT, kD, kD);
  cvtT(Wk, WkbT, kD, kNKV);
  cvtT(Wv, WvbT, kD, kNKV);
  cvtT(Wo, WobT, kD, kD);

  dim3 gQ(kD / 128, kM / 128);     // 16 x 32
  dim3 gKV(kNKV / 128, kM / 128);  // 4 x 32
  gemm_bf16<<<gQ,  256, 0, stream>>>(xb,  WqbT, bq, Qb, nullptr, nullptr, kM, kD,   kD);
  gemm_bf16<<<gKV, 256, 0, stream>>>(kvb, WkbT, bk, Kb, nullptr, nullptr, kM, kNKV, kD);
  gemm_bf16<<<gKV, 256, 0, stream>>>(kvb, WvbT, bv, nullptr, nullptr, VtG, kM, kNKV, kD);

  dim3 gF(kS / 64, kH, kB);        // 32 x 32 x 2
  flash_attn<<<gF, 128, 0, stream>>>(Qb, Kb, VtG, Ob);

  gemm_bf16<<<gQ, 256, 0, stream>>>(Ob, WobT, bo, nullptr, out, nullptr, kM, kD, kD);
}
